// GoMambaSpatialNet_87540023427864
// MI455X (gfx1250) — matrix operand order repaired and checked by reading, not compiled, not run on hardware
//
#include <hip/hip_runtime.h>
#include <hip/hip_bf16.h>
#include <math.h>

typedef _Float16 half_t;
typedef __attribute__((ext_vector_type(16))) _Float16 v16h;
typedef __attribute__((ext_vector_type(8)))  float    v8f;

#define NLAYERS 12
#define DSTATE  16
#define DINNER  1024
#define DTRANK  32
#define SEQ_L   361
#define BATCH   64
#define M_TOK   (BATCH * SEQ_L)   // 23104 = 64*361, divisible by 64

__device__ __forceinline__ float sigmoidf_(float v) { return 1.0f / (1.0f + __expf(-v)); }
__device__ __forceinline__ float siluf_(float v)    { return v * sigmoidf_(v); }

// ---------------------------------------------------------------------------
// WMMA GEMM: C[M,N] (+)= A[M,K] * W[N,K]^T   (W row-major [N,K], f32 weights)
// Block: 256 threads = 8 waves. Tile BM=64 x BN=64, K-chunk 32.
// Wave w -> 16x32 strip: m_sub = w&3, n_half = w>>2; two accumulators share
// one A fragment (2 back-to-back v_wmma_f32_16x16x32_f16 per chunk).
// global_prefetch_b8 issued for the next K-chunk while this one computes.
// Requires M%64==0, N%64==0, K%32==0 (true for all call sites here).
// ---------------------------------------------------------------------------
template <typename TA, typename TC>
__global__ void __launch_bounds__(256)
wmma_gemm(const TA* __restrict__ A, int lda,
          const float* __restrict__ W,
          TC* __restrict__ C, int ldc,
          int K, int accumulate)
{
    __shared__ half_t As[64][32];   // 4 KB
    __shared__ half_t Bs[64][32];   // 4 KB  (Bs[n][k] => W-tile transpose is implicit)

    const int tid    = threadIdx.x;
    const int lane   = tid & 31;
    const int wave   = tid >> 5;
    const int m_sub  = wave & 3;          // 0..3  -> rows  m_sub*16 .. +15
    const int n_half = wave >> 2;         // 0..1  -> cols  n_half*32 .. +31
    const int mBase  = blockIdx.x * 64;
    const int nBase  = blockIdx.y * 64;

    v8f acc0 = {}, acc1 = {};

    const int rowT = tid >> 2;            // 0..63
    const int kqT  = (tid & 3) * 8;       // 0,8,16,24  (8 contiguous elems/thread)

    for (int k0 = 0; k0 < K; k0 += 32) {
        // Stage A tile 64x32 (convert to f16 if needed)
        {
            const TA* src = A + (long)(mBase + rowT) * lda + k0 + kqT;
#pragma unroll
            for (int i = 0; i < 8; ++i) As[rowT][kqT + i] = (half_t)(float)src[i];
            if (k0 + 32 < K) __builtin_prefetch(src + 32, 0, 1);
        }
        // Stage W tile 64x32 (f32 -> f16)
        {
            const float* src = W + (long)(nBase + rowT) * K + k0 + kqT;
#pragma unroll
            for (int i = 0; i < 8; ++i) Bs[rowT][kqT + i] = (half_t)src[i];
            if (k0 + 32 < K) __builtin_prefetch(src + 32, 0, 1);
        }
        __syncthreads();

        // Fragment gathers (ISA 7.12.2):
        // A 16x32: M = lane&15; half i -> K = i + (i&8) + (lane>=16 ? 8 : 0)
        // B 32x16: N = lane&15; half i -> K = i + (i&8) + (lane>=16 ? 8 : 0)
        v16h av, bv0, bv1;
        const int mrow  = (m_sub << 4) + (lane & 15);
        const int nrow0 = (n_half << 5) + (lane & 15);
        const int nrow1 = nrow0 + 16;
        const int ahi   = (lane >> 4) << 3;   // 0 or 8

#pragma unroll
        for (int i = 0; i < 16; ++i) {
            av[i]  = As[mrow][i + (i & 8) + ahi];
            bv0[i] = Bs[nrow0][i + (i & 8) + ahi];
            bv1[i] = Bs[nrow1][i + (i & 8) + ahi];
        }
        acc0 = __builtin_amdgcn_wmma_f32_16x16x32_f16(
            false, av, false, bv0, (short)0, acc0, false, false);
        acc1 = __builtin_amdgcn_wmma_f32_16x16x32_f16(
            false, av, false, bv1, (short)0, acc1, false, false);
        __syncthreads();
    }

    // C/D layout: VGPR r -> M = r + (lane>=16 ? 8 : 0); N = lane&15
    const int nCol0 = nBase + (n_half << 5) + (lane & 15);
    const int mRow0 = mBase + (m_sub << 4) + ((lane >> 4) << 3);
#pragma unroll
    for (int r = 0; r < 8; ++r) {
        const long i0 = (long)(mRow0 + r) * ldc + nCol0;
        const long i1 = i0 + 16;
        float v0 = acc0[r];
        float v1 = acc1[r];
        if (accumulate) { v0 += (float)C[i0]; v1 += (float)C[i1]; }
        C[i0] = (TC)v0;
        C[i1] = (TC)v1;
    }
}

// ---------------------------------------------------------------------------
// Conv stem: 3x3 conv 17->128 + bias + SiLU, written NHWC f16 (so 1x1 conv2
// becomes a plain WMMA GEMM with K=128).
// ---------------------------------------------------------------------------
__global__ void __launch_bounds__(256)
conv1_kernel(const float* __restrict__ x, const float* __restrict__ w,
             const float* __restrict__ bias, half_t* __restrict__ out)
{
    const int idx = blockIdx.x * blockDim.x + threadIdx.x;
    if (idx >= M_TOK * 128) return;
    const int co  = idx & 127;
    const int pos = (idx >> 7) % SEQ_L;
    const int b   = idx / (128 * SEQ_L);
    const int py = pos / 19, px = pos % 19;
    float acc = bias[co];
    for (int ci = 0; ci < 17; ++ci) {
        const float* xin = x + (long)(b * 17 + ci) * 361;
        const float* wf  = w + (long)(co * 17 + ci) * 9;
#pragma unroll
        for (int ky = 0; ky < 3; ++ky) {
            const int yy = py + ky - 1;
            if (yy < 0 || yy >= 19) continue;
#pragma unroll
            for (int kx = 0; kx < 3; ++kx) {
                const int xx = px + kx - 1;
                if (xx < 0 || xx >= 19) continue;
                acc += xin[yy * 19 + xx] * wf[ky * 3 + kx];
            }
        }
    }
    out[idx] = (half_t)siluf_(acc);
}

// u[m,512] += conv2_b[e] + spatial_link[l,e]
__global__ void __launch_bounds__(256)
bias_link_kernel(float* __restrict__ u, const float* __restrict__ b2,
                 const float* __restrict__ slink)
{
    const int idx = blockIdx.x * blockDim.x + threadIdx.x;
    if (idx >= M_TOK * 512) return;
    const int e = idx & 511;
    const int l = (idx >> 9) % SEQ_L;
    u[idx] += b2[e] + slink[l * 512 + e];
}

// Causal depthwise conv over L (taps=4) + bias + SiLU. Reads xc half of xz.
__global__ void __launch_bounds__(256)
dwconv_kernel(const half_t* __restrict__ xz, const float* __restrict__ cw,
              const float* __restrict__ cb, half_t* __restrict__ xc)
{
    const int idx = blockIdx.x * blockDim.x + threadIdx.x;
    if (idx >= M_TOK * DINNER) return;
    const int d = idx & 1023;
    const int m = idx >> 10;
    const int l = m % SEQ_L;
    const int base = m - l;                 // b*361
    float acc = cb[d];
#pragma unroll
    for (int k = 0; k < 4; ++k) {
        const int lk = l - 3 + k;
        if (lk >= 0)
            acc += (float)xz[(long)(base + lk) * 2048 + d] * cw[d * 4 + k];
    }
    xc[idx] = (half_t)siluf_(acc);
}

// dt = softplus(dt_raw + dt_proj_b), in place on f16 buffer
__global__ void __launch_bounds__(256)
softplus_kernel(half_t* __restrict__ dt, const float* __restrict__ dpb)
{
    const int idx = blockIdx.x * blockDim.x + threadIdx.x;
    if (idx >= M_TOK * DINNER) return;
    const float v = (float)dt[idx] + dpb[idx & 1023];
    dt[idx] = (half_t)(v > 20.0f ? v : log1pf(__expf(v)));
}

// Selective scan: thread = (b,d) channel, 16-state recurrence in registers.
// Fuses D_skip and z-gating; B_t/C_t broadcast per step via LDS.
__global__ void __launch_bounds__(256)
scan_kernel(const half_t* __restrict__ dt, const float* __restrict__ dbc,
            const half_t* __restrict__ xc, const half_t* __restrict__ xz,
            const float* __restrict__ alog, const float* __restrict__ dsk,
            half_t* __restrict__ ybuf)
{
    __shared__ float BC[32];                // [0:16)=B_t, [16:32)=C_t
    const int b = blockIdx.x >> 2;
    const int d = ((blockIdx.x & 3) << 8) + threadIdx.x;

    float a[DSTATE], h[DSTATE];
#pragma unroll
    for (int n = 0; n < DSTATE; ++n) {
        a[n] = -__expf(alog[d * DSTATE + n]);
        h[n] = 0.0f;
    }
    const float skip = dsk[d];

    for (int l = 0; l < SEQ_L; ++l) {
        const long row = (long)(b * SEQ_L + l);
        __syncthreads();
        if (threadIdx.x < 32) BC[threadIdx.x] = dbc[row * 64 + 32 + threadIdx.x];
        __syncthreads();

        const float dtv = (float)dt[row * 1024 + d];
        const float ut  = (float)xc[row * 1024 + d];
        const float du  = dtv * ut;
        float y = 0.0f;
#pragma unroll
        for (int n = 0; n < DSTATE; ++n) {
            h[n] = __expf(dtv * a[n]) * h[n] + du * BC[n];
            y += h[n] * BC[16 + n];
        }
        y += ut * skip;
        const float zv = (float)xz[row * 2048 + 1024 + d];
        y *= siluf_(zv);
        ybuf[row * 1024 + d] = (half_t)y;
    }
}

__global__ void __launch_bounds__(256)
zero_kernel(float* __restrict__ p, int n)
{
    const int idx = blockIdx.x * blockDim.x + threadIdx.x;
    if (idx < n) p[idx] = 0.0f;
}

// LayerNorm over D=512 per token, accumulate mean over L into xg[b,512]
__global__ void __launch_bounds__(256)
lnmean_kernel(const float* __restrict__ u, const float* __restrict__ g,
              const float* __restrict__ bt, float* __restrict__ xg)
{
    __shared__ float s1[256], s2[256];
    const int m = blockIdx.x;
    const int t = threadIdx.x;
    const float* row = u + (long)m * 512;
    const float a0 = row[t], a1 = row[t + 256];
    s1[t] = a0 + a1;
    s2[t] = a0 * a0 + a1 * a1;
    __syncthreads();
    for (int o = 128; o > 0; o >>= 1) {
        if (t < o) { s1[t] += s1[t + o]; s2[t] += s2[t + o]; }
        __syncthreads();
    }
    const float mu  = s1[0] * (1.0f / 512.0f);
    const float var = s2[0] * (1.0f / 512.0f) - mu * mu;
    const float inv = rsqrtf(var + 1e-5f);
    const int b = m / SEQ_L;
    atomicAdd(&xg[b * 512 + t],       ((a0 - mu) * inv * g[t]       + bt[t])       * (1.0f / 361.0f));
    atomicAdd(&xg[b * 512 + t + 256], ((a1 - mu) * inv * g[t + 256] + bt[t + 256]) * (1.0f / 361.0f));
}

// Policy (362 logits) + value head (tanh). One block per batch element.
__global__ void __launch_bounds__(256)
heads_kernel(const float* __restrict__ xg, const float* __restrict__ pw,
             const float* __restrict__ pb, const float* __restrict__ vw,
             const float* __restrict__ vb, float* __restrict__ out)
{
    __shared__ float xs[512];
    const int b = blockIdx.x, t = threadIdx.x;
    xs[t] = xg[b * 512 + t];
    xs[t + 256] = xg[b * 512 + t + 256];
    __syncthreads();
    for (int o = t; o < 362; o += 256) {
        float acc = pb[o];
        const float* wr = pw + (long)o * 512;
        for (int k = 0; k < 512; ++k) acc += xs[k] * wr[k];
        out[b * 362 + o] = acc;
    }
    if (t == 0) {
        float acc = vb[0];
        for (int k = 0; k < 512; ++k) acc += xs[k] * vw[k];
        out[BATCH * 362 + b] = tanhf(acc);
    }
}

// ---------------------------------------------------------------------------
extern "C" void kernel_launch(void* const* d_in, const int* in_sizes, int n_in,
                              void* d_out, int out_size, void* d_ws, size_t ws_size,
                              hipStream_t stream)
{
    const float* x        = (const float*)d_in[0];
    const float* c1w      = (const float*)d_in[1];
    const float* c1b      = (const float*)d_in[2];
    const float* c2w      = (const float*)d_in[3];
    const float* c2b      = (const float*)d_in[4];
    const float* slink    = (const float*)d_in[5];
    const float* ipw_all  = (const float*)d_in[6];
    const float* cw_all   = (const float*)d_in[7];
    const float* cb_all   = (const float*)d_in[8];
    const float* xpw_all  = (const float*)d_in[9];
    const float* dpw_all  = (const float*)d_in[10];
    const float* dpb_all  = (const float*)d_in[11];
    const float* alog_all = (const float*)d_in[12];
    const float* dsk_all  = (const float*)d_in[13];
    const float* opw_all  = (const float*)d_in[14];
    const float* ln_g     = (const float*)d_in[15];
    const float* ln_b     = (const float*)d_in[16];
    const float* pw       = (const float*)d_in[17];
    const float* pb       = (const float*)d_in[18];
    const float* vw       = (const float*)d_in[19];
    const float* vb       = (const float*)d_in[20];
    (void)in_sizes; (void)n_in; (void)out_size; (void)ws_size;

    char* ws = (char*)d_ws;
    size_t off = 0;
    auto alloc = [&](size_t bytes) -> void* {
        void* p = ws + off;
        off += (bytes + 255) & ~(size_t)255;
        return p;
    };
    float*  u    = (float*) alloc((size_t)M_TOK * 512  * sizeof(float));   // residual stream
    half_t* xzb  = (half_t*)alloc((size_t)M_TOK * 2048 * sizeof(half_t));  // in_proj out
    half_t* xcb  = (half_t*)alloc((size_t)M_TOK * 1024 * sizeof(half_t));  // conv1d+silu
    float*  dbcb = (float*) alloc((size_t)M_TOK * 64   * sizeof(float));   // x_proj out
    half_t* dtb  = (half_t*)alloc((size_t)M_TOK * 1024 * sizeof(half_t));  // dt
    half_t* yb   = (half_t*)alloc((size_t)M_TOK * 1024 * sizeof(half_t));  // gated scan out
    half_t* hb   = (half_t*)alloc((size_t)M_TOK * 128  * sizeof(half_t));  // conv1 out (NHWC)
    float*  xg   = (float*) alloc((size_t)BATCH * 512  * sizeof(float));

    const int EW = 256;

    // Stem
    conv1_kernel<<<(M_TOK * 128 + EW - 1) / EW, EW, 0, stream>>>(x, c1w, c1b, hb);
    wmma_gemm<half_t, float><<<dim3(M_TOK / 64, 512 / 64), 256, 0, stream>>>(
        hb, 128, c2w, u, 512, /*K=*/128, 0);
    bias_link_kernel<<<(M_TOK * 512 + EW - 1) / EW, EW, 0, stream>>>(u, c2b, slink);

    // Mamba layers (residual)
    for (int layer = 0; layer < NLAYERS; ++layer) {
        const float* ipw  = ipw_all  + (size_t)layer * 2048 * 512;
        const float* cwL  = cw_all   + (size_t)layer * 1024 * 4;
        const float* cbL  = cb_all   + (size_t)layer * 1024;
        const float* xpw  = xpw_all  + (size_t)layer * 64 * 1024;
        const float* dpw  = dpw_all  + (size_t)layer * 1024 * 32;
        const float* dpbL = dpb_all  + (size_t)layer * 1024;
        const float* alog = alog_all + (size_t)layer * 1024 * 16;
        const float* dskL = dsk_all  + (size_t)layer * 1024;
        const float* opw  = opw_all  + (size_t)layer * 512 * 1024;

        // xz = u @ in_proj^T   [23104,512] x [2048,512]^T
        wmma_gemm<float, half_t><<<dim3(M_TOK / 64, 2048 / 64), 256, 0, stream>>>(
            u, 512, ipw, xzb, 2048, /*K=*/512, 0);
        // depthwise causal conv + silu on xc half
        dwconv_kernel<<<(M_TOK * 1024 + EW - 1) / EW, EW, 0, stream>>>(xzb, cwL, cbL, xcb);
        // dbc = xc @ x_proj^T  [23104,1024] x [64,1024]^T
        wmma_gemm<half_t, float><<<dim3(M_TOK / 64, 64 / 64), 256, 0, stream>>>(
            xcb, 1024, xpw, dbcb, 64, /*K=*/1024, 0);
        // dt_raw = dt_r @ dt_proj^T  (dt_r = dbc[:, :32], lda = 64)
        wmma_gemm<float, half_t><<<dim3(M_TOK / 64, 1024 / 64), 256, 0, stream>>>(
            dbcb, 64, dpw, dtb, 1024, /*K=*/32, 0);
        softplus_kernel<<<(M_TOK * 1024 + EW - 1) / EW, EW, 0, stream>>>(dtb, dpbL);
        // selective scan + D_skip + z-gate
        scan_kernel<<<BATCH * 4, 256, 0, stream>>>(dtb, dbcb, xcb, xzb, alog, dskL, yb);
        // u += y @ out_proj^T  [23104,1024] x [512,1024]^T (accumulate)
        wmma_gemm<half_t, float><<<dim3(M_TOK / 64, 512 / 64), 256, 0, stream>>>(
            yb, 1024, opw, u, 512, /*K=*/1024, 1);
    }

    // LN -> mean over L -> heads
    zero_kernel<<<(BATCH * 512 + EW - 1) / EW, EW, 0, stream>>>(xg, BATCH * 512);
    lnmean_kernel<<<M_TOK, 256, 0, stream>>>(u, ln_g, ln_b, xg);
    heads_kernel<<<BATCH, 256, 0, stream>>>(xg, pw, pb, vw, vb, (float*)d_out);
}
